// PhysaNetLayer_41394894799174
// MI455X (gfx1250) — matrix operand order repaired and checked by reading, not compiled, not run on hardware
//
#include <hip/hip_runtime.h>
#include <hip/hip_bf16.h>
#include <math.h>

// Problem constants (from reference)
#define Bc 4
#define Nc 20000
#define Ec 160000
#define DN 128
#define DE 32
#define DH 128
#define KPAD 320   // msg_in = 289 padded to 320 (multiple of 32)
#define KIN2 256   // node-MLP input = 128 (LN feat) + 128 (agg)

typedef __attribute__((ext_vector_type(16))) __bf16          v16bf;
typedef __attribute__((ext_vector_type(16))) unsigned short  v16u;
typedef __attribute__((ext_vector_type(8)))  unsigned short  v8u;
typedef __attribute__((ext_vector_type(4)))  unsigned short  v4u;
typedef __attribute__((ext_vector_type(8)))  float           v8f;

// ---------------- bf16 helpers (round-to-nearest-even) ----------------
__device__ __forceinline__ unsigned short f2bf(float f) {
  unsigned int u = __float_as_uint(f);
  u += 0x7FFFu + ((u >> 16) & 1u);
  return (unsigned short)(u >> 16);
}
__device__ __forceinline__ float bf2f(unsigned short h) {
  return __uint_as_float(((unsigned int)h) << 16);
}
__device__ __forceinline__ void st_bf4(unsigned short* dst, float4 v) {
  v4u p; p[0] = f2bf(v.x); p[1] = f2bf(v.y); p[2] = f2bf(v.z); p[3] = f2bf(v.w);
  *(v4u*)dst = p;   // 8-byte LDS store
}
__device__ __forceinline__ v16bf make_frag(const unsigned short* p0,
                                           const unsigned short* p1) {
  v8u lo = *(const v8u*)p0;     // ds_load_b128
  v8u hi = *(const v8u*)p1;     // ds_load_b128
  v16u t;
#pragma unroll
  for (int i = 0; i < 8; ++i) { t[i] = lo[i]; t[i + 8] = hi[i]; }
  return __builtin_bit_cast(v16bf, t);
}
__device__ __forceinline__ float gelu_exact(float x) {
  return 0.5f * x * (1.0f + erff(x * 0.70710678118654752f));
}

// ---------------- WMMA row-strip GEMM ----------------
// One wave computes rows [16w,16w+16) x all 128 cols of a 128x128 output tile.
// A in LDS row-major (M x K, bf16), B in LDS N-major (N x K, bf16, i.e. W^T),
// accumulators initialized from bias (per-column).
// Layouts per CDNA5 ISA 7.12.2:
//   A 16x32 frag: lane&15 = M row; lanes<16 hold K {0..7,16..23}, lanes>=16 {8..15,24..31}
//   B 32x16 frag: lane&15 = N col; lanes<16 hold K {0..7,16..23}, lanes>=16 {8..15,24..31}
//   C/D:          VGPR r -> M = r (lanes<16) / 8+r (lanes>=16); N = lane&15
template <int KSZ>
__device__ __forceinline__ void gemm_rowstrip(const unsigned short* __restrict__ sA,
                                              int lda,
                                              const unsigned short* __restrict__ sBt,
                                              int ldb,
                                              const float* __restrict__ bias,
                                              v8f acc[8], int w, int lane) {
  const int cl  = lane & 15;
  const int kgb = (lane < 16) ? 0 : 8;
  const int khf = (lane < 16) ? 0 : 16;
#pragma unroll
  for (int t = 0; t < 8; ++t) {
    float bv = bias[t * 16 + cl];
#pragma unroll
    for (int i = 0; i < 8; ++i) acc[t][i] = bv;
  }
  const unsigned short* Arow = sA + (16 * w + cl) * lda + kgb;
#pragma unroll
  for (int kb = 0; kb < KSZ; kb += 32) {
    v16bf a = make_frag(Arow + kb, Arow + kb + 16);
#pragma unroll
    for (int t = 0; t < 8; ++t) {
      const unsigned short* bp = sBt + (t * 16 + cl) * ldb + kb + kgb;
      v16bf b = make_frag(bp, bp + 16);
      acc[t] = __builtin_amdgcn_wmma_f32_16x16x32_bf16(
          false, a, false, b, (short)0, acc[t], false, false);
    }
  }
}

__device__ __forceinline__ void store_gelu_bf16(const v8f acc[8],
                                                unsigned short* sH, int w, int lane) {
  const int cl = lane & 15;
  const int ro = (lane < 16) ? 0 : 8;
#pragma unroll
  for (int t = 0; t < 8; ++t)
#pragma unroll
    for (int r = 0; r < 8; ++r)
      sH[(16 * w + ro + r) * 128 + t * 16 + cl] = f2bf(gelu_exact(acc[t][r]));
}
__device__ __forceinline__ void store_f32(const v8f acc[8], float* sM, int w, int lane) {
  const int cl = lane & 15;
  const int ro = (lane < 16) ? 0 : 8;
#pragma unroll
  for (int t = 0; t < 8; ++t)
#pragma unroll
    for (int r = 0; r < 8; ++r)
      sM[(16 * w + ro + r) * 128 + t * 16 + cl] = acc[t][r];
}

// ---------------- kernel: zero scratch ----------------
__global__ void k_zero(float4* __restrict__ p, int n4) {
  int i = blockIdx.x * blockDim.x + threadIdx.x;
  if (i < n4) p[i] = make_float4(0.f, 0.f, 0.f, 0.f);
}

// ---------------- kernel: transpose weights to bf16 (N-major, K padded) ----
__global__ void k_prep_weights(const float* __restrict__ W1, const float* __restrict__ W2,
                               const float* __restrict__ Wn1, const float* __restrict__ Wn2,
                               unsigned short* __restrict__ W1T, unsigned short* __restrict__ W2T,
                               unsigned short* __restrict__ Wn1T, unsigned short* __restrict__ Wn2T) {
  int i = blockIdx.x * 256 + threadIdx.x;
  if (i < 128 * KPAD) {                // W1: (289,128) -> (128,320) padded
    int n = i / KPAD, k = i - n * KPAD;
    W1T[i] = (k < 289) ? f2bf(W1[k * 128 + n]) : (unsigned short)0;
    return;
  }
  i -= 128 * KPAD;
  if (i < 128 * 128) {                 // W2: (128,128) -> (128,128)
    int n = i >> 7, k = i & 127;
    W2T[i] = f2bf(W2[k * 128 + n]);
    return;
  }
  i -= 128 * 128;
  if (i < 128 * KIN2) {                // Wn1: (256,128) -> (128,256)
    int n = i / KIN2, k = i - n * KIN2;
    Wn1T[i] = f2bf(Wn1[k * 128 + n]);
    return;
  }
  i -= 128 * KIN2;
  if (i < 128 * 128) {                 // Wn2: (128,128) -> (128,128)
    int n = i >> 7, k = i & 127;
    Wn2T[i] = f2bf(Wn2[k * 128 + n]);
  }
}

// ---------------- kernel 1: edge MLP + Q + plasticity + scatter-max -------
// One block = 128 edge rows, 256 threads (8 wave32).
// Dynamic LDS: A(128x320 bf16) | W1T(128x320 bf16) | W2T(128x128 bf16)
//            | H(128x128 bf16) | MSG(128x128 f32)  = 294912 B (<320KB/WGP)
__global__ __launch_bounds__(256) void k_edge_mlp(
    const float* __restrict__ node_feat, const float* __restrict__ edge_feat,
    const float* __restrict__ edge_w,
    const int* __restrict__ src, const int* __restrict__ dst,
    const unsigned short* __restrict__ W1T, const unsigned short* __restrict__ W2T,
    const float* __restrict__ b1, const float* __restrict__ b2,
    const float* __restrict__ alpha_p, const float* __restrict__ mu_p,
    const float* __restrict__ beta_p,
    float* __restrict__ edge_w_out, unsigned short* __restrict__ msgs,
    float* __restrict__ node_max) {
  extern __shared__ char smem[];
  unsigned short* sA  = (unsigned short*)smem;        // 128*320
  unsigned short* sW1 = sA + 128 * KPAD;              // 128*320
  unsigned short* sW2 = sW1 + 128 * KPAD;             // 128*128
  unsigned short* sH  = sW2 + 128 * 128;              // 128*128
  float*          sM  = (float*)(sH + 128 * 128);     // 128*128 f32

  const int tid = threadIdx.x, lane = tid & 31, w = tid >> 5;
  const int rowBase = blockIdx.x * 128;               // over B*E rows

  __builtin_prefetch(W1T, 0, 3);
  __builtin_prefetch(W2T, 0, 3);

  // ---- gather + bf16-convert A tile: 2 threads per edge row ----
  {
    const int r = tid >> 1, half = tid & 1;
    const int g = rowBase + r;
    const int b = g / Ec, e = g - b * Ec;
    const float4* hs = (const float4*)(node_feat + ((size_t)b * Nc + src[e]) * DN);
    const float4* hd = (const float4*)(node_feat + ((size_t)b * Nc + dst[e]) * DN);
    const float4* ef = (const float4*)(edge_feat + ((size_t)b * Ec + e) * DE);
    unsigned short* Ar = sA + r * KPAD;
    if (!half) {
#pragma unroll
      for (int c = 0; c < 32; ++c) st_bf4(Ar + 4 * c, hs[c]);          // h_src
#pragma unroll
      for (int c = 0; c < 8; ++c)  st_bf4(Ar + 128 + 4 * c, hd[c]);    // h_dst[0:32)
    } else {
#pragma unroll
      for (int c = 8; c < 32; ++c) st_bf4(Ar + 128 + 4 * c, hd[c]);    // h_dst[32:128)
#pragma unroll
      for (int c = 0; c < 8; ++c)  st_bf4(Ar + 256 + 4 * c, ef[c]);    // edge_feat
      Ar[288] = f2bf(edge_w[(size_t)b * Ec + e]);                      // edge_w
#pragma unroll
      for (int c = 289; c < KPAD; ++c) Ar[c] = 0;                      // K pad
    }
  }
  // ---- stage weights (coalesced 16B copies) ----
  for (int i = tid; i < (128 * KPAD) / 8; i += 256) ((v8u*)sW1)[i] = ((const v8u*)W1T)[i];
  for (int i = tid; i < (128 * 128) / 8; i += 256)  ((v8u*)sW2)[i] = ((const v8u*)W2T)[i];
  __syncthreads();

  v8f acc[8];
  gemm_rowstrip<KPAD>(sA, KPAD, sW1, KPAD, b1, acc, w, lane);  // (128x320)@(320x128)
  store_gelu_bf16(acc, sH, w, lane);
  __syncthreads();
  gemm_rowstrip<128>(sH, 128, sW2, 128, b2, acc, w, lane);     // (128x128)@(128x128)
  store_f32(acc, sM, w, lane);
  __syncthreads();

  // ---- messages -> global scratch as bf16 (coalesced) ----
  for (int base = 0; base < 128 * 128; base += 256 * 8) {
    const int i = base + tid * 8;
    v8u p;
#pragma unroll
    for (int j = 0; j < 8; ++j) p[j] = f2bf(sM[i + j]);
    *(v8u*)(msgs + (size_t)rowBase * DH + i) = p;
  }

  // ---- per-edge Q, plasticity update, scatter-max for softmax ----
  if (tid < 128) {
    const int r = tid;
    float s = 0.f;
#pragma unroll 8
    for (int cc = 0; cc < 128; ++cc) {       // staggered -> LDS-bank friendly
      const int c = (cc + r) & 127;
      s += fabsf(sM[r * 128 + c]);
    }
    const float Q = s * (1.0f / 128.0f);
    const int g = rowBase + r;
    const int b = g / Ec, e = g - b * Ec;
    const float ew    = edge_w[(size_t)b * Ec + e];
    const float alpha = alpha_p[0], mu = mu_p[0], beta = beta_p[0];
    const float mu_c  = fminf(fmaxf(mu, 0.1f), 3.0f);
    const float Qm    = powf(fmaxf(Q, 1e-8f), mu_c);
    const float wn    = fminf(fmaxf(ew + alpha * (Qm - beta * ew), 1e-6f), 10.0f);
    edge_w_out[g] = wn;
    // wn > 0 and node_max init 0 -> int-compare == float-compare
    atomicMax((int*)(node_max + (size_t)b * Nc + src[e]), __float_as_int(wn));
  }
}

// ---------------- kernel 2: exp(w - max[src]) + scatter-add of sums -------
__global__ void k_softmax_sum(const float* __restrict__ edge_w_new,
                              const int* __restrict__ src,
                              const float* __restrict__ node_max,
                              float* __restrict__ w_exp,
                              float* __restrict__ node_sum) {
  const int i = blockIdx.x * blockDim.x + threadIdx.x;
  if (i >= Bc * Ec) return;
  const int b = i / Ec, e = i - b * Ec;
  const int s = src[e];
  const float we = expf(edge_w_new[i] - node_max[(size_t)b * Nc + s]);
  w_exp[i] = we;
  atomicAdd(node_sum + (size_t)b * Nc + s, we);
}

// ---------------- kernel 3: normalized scatter-add of messages ------------
// 256 threads = 2 edges x 128 channels.
__global__ __launch_bounds__(256) void k_scatter_agg(
    const float* __restrict__ w_exp, const float* __restrict__ node_sum,
    const int* __restrict__ src, const int* __restrict__ dst,
    const unsigned short* __restrict__ msgs, float* __restrict__ agg) {
  const int i = blockIdx.x * 2 + (threadIdx.x >> 7);
  const int c = threadIdx.x & 127;
  if (i >= Bc * Ec) return;
  const int b = i / Ec, e = i - b * Ec;
  const float nrm = w_exp[i] / (node_sum[(size_t)b * Nc + src[e]] + 1e-8f);
  const float m = bf2f(msgs[(size_t)i * DH + c]);
  atomicAdd(agg + ((size_t)b * Nc + dst[e]) * DH + c, nrm * m);
}

// ---------------- kernel 4: node MLP (LN -> GEMM -> GELU -> GEMM -> LN) ---
// One block = 128 node rows, 256 threads (8 wave32).
// Dynamic LDS: X(128x256 bf16) | Wn1T(128x256 bf16) | Wn2T(128x128 bf16)
//            | H(128x128 bf16) | OUT(128x128 f32) = 262144 B
__global__ __launch_bounds__(256) void k_node_mlp(
    const float* __restrict__ node_feat, const float* __restrict__ agg,
    const unsigned short* __restrict__ Wn1T, const unsigned short* __restrict__ Wn2T,
    const float* __restrict__ bn1, const float* __restrict__ bn2,
    const float* __restrict__ g1, const float* __restrict__ bta1,
    const float* __restrict__ g2, const float* __restrict__ bta2,
    float* __restrict__ node_out) {
  extern __shared__ char smem[];
  unsigned short* sX  = (unsigned short*)smem;        // 128*256
  unsigned short* sW1 = sX + 128 * KIN2;              // 128*256
  unsigned short* sW2 = sW1 + 128 * KIN2;             // 128*128
  unsigned short* sH  = sW2 + 128 * 128;              // 128*128
  float*          sO  = (float*)(sH + 128 * 128);     // 128*128 f32

  const int tid = threadIdx.x, lane = tid & 31, w = tid >> 5;
  const int rowBase = blockIdx.x * 128;               // over B*N rows

  // ---- stage X = [LN(node_feat), agg] as bf16: 2 threads per row ----
  {
    const int r = tid >> 1, half = tid & 1;
    const int g = rowBase + r;
    unsigned short* Xr = sX + r * KIN2;
    if (!half) {
      const float* nf = node_feat + (size_t)g * DN;
      float s = 0.f, s2 = 0.f;
#pragma unroll 8
      for (int cc = 0; cc < 128; ++cc) {
        const int c = (cc + r) & 127;
        const float x = nf[c];
        s += x; s2 += x * x;
      }
      const float m  = s * (1.0f / 128.0f);
      const float v  = s2 * (1.0f / 128.0f) - m * m;
      const float rs = rsqrtf(v + 1e-5f);
#pragma unroll 4
      for (int c = 0; c < 128; ++c)
        Xr[c] = f2bf((nf[c] - m) * rs * g1[c] + bta1[c]);
    } else {
      const float4* ag = (const float4*)(agg + (size_t)g * DH);
#pragma unroll
      for (int c = 0; c < 32; ++c) st_bf4(Xr + 128 + 4 * c, ag[c]);
    }
  }
  for (int i = tid; i < (128 * KIN2) / 8; i += 256) ((v8u*)sW1)[i] = ((const v8u*)Wn1T)[i];
  for (int i = tid; i < (128 * 128) / 8; i += 256)  ((v8u*)sW2)[i] = ((const v8u*)Wn2T)[i];
  __syncthreads();

  v8f acc[8];
  gemm_rowstrip<KIN2>(sX, KIN2, sW1, KIN2, bn1, acc, w, lane);  // (128x256)@(256x128)
  store_gelu_bf16(acc, sH, w, lane);
  __syncthreads();
  gemm_rowstrip<128>(sH, 128, sW2, 128, bn2, acc, w, lane);     // (128x128)@(128x128)
  store_f32(acc, sO, w, lane);
  __syncthreads();

  // ---- residual + final LayerNorm + write out ----
  if (tid < 128) {
    const int r = tid;
    const int g = rowBase + r;
    const float* nf = node_feat + (size_t)g * DN;
    float s = 0.f, s2 = 0.f;
#pragma unroll 8
    for (int cc = 0; cc < 128; ++cc) {
      const int c = (cc + r) & 127;
      const float y = nf[c] + sO[r * 128 + c];
      s += y; s2 += y * y;
    }
    const float m  = s * (1.0f / 128.0f);
    const float v  = s2 * (1.0f / 128.0f) - m * m;
    const float rs = rsqrtf(v + 1e-5f);
    float* out = node_out + (size_t)g * DN;
#pragma unroll 4
    for (int c = 0; c < 128; c += 4) {
      float4 o;
      o.x = (nf[c + 0] + sO[r * 128 + c + 0] - m) * rs * g2[c + 0] + bta2[c + 0];
      o.y = (nf[c + 1] + sO[r * 128 + c + 1] - m) * rs * g2[c + 1] + bta2[c + 1];
      o.z = (nf[c + 2] + sO[r * 128 + c + 2] - m) * rs * g2[c + 2] + bta2[c + 2];
      o.w = (nf[c + 3] + sO[r * 128 + c + 3] - m) * rs * g2[c + 3] + bta2[c + 3];
      *(float4*)(out + c) = o;
    }
  }
}

// ---------------- workspace layout (bytes, all 256-aligned) ----------------
// 0        : W1T  bf16 128x320   ( 81920)
// 81920    : W2T  bf16 128x128   ( 32768)
// 114688   : Wn1T bf16 128x256   ( 65536)
// 180224   : Wn2T bf16 128x128   ( 32768)
// 212992   : node_max f32 B*N    (320000)
// 532992   : node_sum f32 B*N    (320000)
// 852992   : w_exp    f32 B*E    (2560000)
// 3412992  : agg      f32 B*N*128(40960000)
// 44372992 : msgs    bf16 B*E*128(163840000)  -> total ~208.3 MB
extern "C" void kernel_launch(void* const* d_in, const int* in_sizes, int n_in,
                              void* d_out, int out_size, void* d_ws, size_t ws_size,
                              hipStream_t stream) {
  const float* node_feat = (const float*)d_in[0];
  const float* edge_feat = (const float*)d_in[1];
  const float* edge_w    = (const float*)d_in[2];
  const int*   src       = (const int*)d_in[3];
  const int*   dst       = (const int*)d_in[4];
  const float* W1  = (const float*)d_in[5];
  const float* b1  = (const float*)d_in[6];
  const float* W2  = (const float*)d_in[7];
  const float* b2  = (const float*)d_in[8];
  const float* Wn1 = (const float*)d_in[9];
  const float* bn1 = (const float*)d_in[10];
  const float* Wn2 = (const float*)d_in[11];
  const float* bn2 = (const float*)d_in[12];
  const float* alpha = (const float*)d_in[13];
  const float* mu    = (const float*)d_in[14];
  const float* beta  = (const float*)d_in[15];
  const float* g1   = (const float*)d_in[16];
  const float* bta1 = (const float*)d_in[17];
  const float* g2   = (const float*)d_in[18];
  const float* bta2 = (const float*)d_in[19];

  char* ws = (char*)d_ws;
  unsigned short* W1T  = (unsigned short*)(ws + 0);
  unsigned short* W2T  = (unsigned short*)(ws + 81920);
  unsigned short* Wn1T = (unsigned short*)(ws + 114688);
  unsigned short* Wn2T = (unsigned short*)(ws + 180224);
  float* node_max = (float*)(ws + 212992);
  float* node_sum = (float*)(ws + 532992);
  float* w_exp    = (float*)(ws + 852992);
  float* agg      = (float*)(ws + 3412992);
  unsigned short* msgs = (unsigned short*)(ws + 44372992);

  float* node_out   = (float*)d_out;
  float* edge_w_out = node_out + (size_t)Bc * Nc * DN;

  // 1) zero node_max, node_sum, w_exp, agg (one contiguous region)
  {
    const int n4 = (44372992 - 212992) / 16;
    k_zero<<<(n4 + 255) / 256, 256, 0, stream>>>((float4*)(ws + 212992), n4);
  }
  // 2) bf16/transpose weight prep
  {
    const int tot = 128 * KPAD + 128 * 128 + 128 * KIN2 + 128 * 128;  // 106496
    k_prep_weights<<<(tot + 255) / 256, 256, 0, stream>>>(W1, W2, Wn1, Wn2,
                                                          W1T, W2T, Wn1T, Wn2T);
  }
  // 3) edge MLP + Q + plasticity + scatter-max (5000 blocks)
  k_edge_mlp<<<(Bc * Ec) / 128, 256, 294912, stream>>>(
      node_feat, edge_feat, edge_w, src, dst, W1T, W2T, b1, b2,
      alpha, mu, beta, edge_w_out, msgs, node_max);
  // 4) exp + sum
  k_softmax_sum<<<(Bc * Ec + 255) / 256, 256, 0, stream>>>(
      edge_w_out, src, node_max, w_exp, node_sum);
  // 5) normalized message aggregation
  k_scatter_agg<<<(Bc * Ec) / 2, 256, 0, stream>>>(
      w_exp, node_sum, src, dst, msgs, agg);
  // 6) node MLP (625 blocks)
  k_node_mlp<<<(Bc * Nc) / 128, 256, 262144, stream>>>(
      node_feat, agg, Wn1T, Wn2T, bn1, bn2, g1, bta1, g2, bta2, node_out);
}
